// NeuralGCDE_2576980378127
// MI455X (gfx1250) — matrix operand order repaired and checked by reading, not compiled, not run on hardware
//
#include <hip/hip_runtime.h>
#include <hip/hip_bf16.h>
#include <math.h>

#define DEV __device__ __forceinline__

static constexpr int B_ = 64, N_ = 307, C_ = 2, H_ = 64, E_ = 10;
static constexpr int M_  = N_ * B_;   // 19648 rows, (node, batch) major
static constexpr int NP_ = 320;       // node count padded to 16
static constexpr int STEPS = 12;

typedef __attribute__((ext_vector_type(16))) __bf16          v16bf;
typedef __attribute__((ext_vector_type(16))) unsigned short  v16u;
typedef __attribute__((ext_vector_type(8)))  float           v8f;

union FragCast { v16u u; v16bf b; };

DEV unsigned short f2bf(float f) {               // fp32 -> bf16 (RNE)
  unsigned u = __float_as_uint(f);
  unsigned r = ((u >> 16) & 1u) + 0x7fffu;
  return (unsigned short)((u + r) >> 16);
}

DEV float fast_tanh(float x) {
#if __has_builtin(__builtin_amdgcn_tanhf)
  return __builtin_amdgcn_tanhf(x);              // hardware V_TANH_F32
#else
  return tanhf(x);
#endif
}

DEV v8f splat8(float x) {
  v8f c;
  #pragma unroll
  for (int r = 0; r < 8; ++r) c[r] = x;
  return c;
}

DEV v8f wmma_bf16(v16u a, v16u b, v8f c) {
  FragCast A, Bm; A.u = a; Bm.u = b;
  return __builtin_amdgcn_wmma_f32_16x16x32_bf16(false, A.b, false, Bm.b,
                                                 (short)0, c, false, false);
}

// async global->LDS copy of 16B per lane (ASYNCcnt-tracked, cdna5_isa 08 §4)
DEV void async_ld_b128(unsigned lds_off, const unsigned short* gsrc) {
  asm volatile("global_load_async_to_lds_b128 %0, %1, off"
               :: "v"(lds_off), "v"((unsigned long long)(size_t)gsrc)
               : "memory");
}

// A fragment (16x32 bf16) from row-major src: lane m=lane&15, half=lane>>4,
// element j -> K = 16*half + j (matches B sequential layout)
DEV v16u gather_a(const unsigned short* base, int stride, int lane) {
  int m = lane & 15, half = lane >> 4;
  const unsigned* rp = (const unsigned*)(base + m * stride);
  v16u f;
  #pragma unroll
  for (int j2 = 0; j2 < 4; ++j2) {
    unsigned d = rp[8 * half + j2];
    f[2*j2]     = (unsigned short)(d & 0xffffu);
    f[2*j2 + 1] = (unsigned short)(d >> 16);
    unsigned e = rp[8 * half + 4 + j2];
    f[8 + 2*j2]     = (unsigned short)(e & 0xffffu);
    f[8 + 2*j2 + 1] = (unsigned short)(e >> 16);
  }
  return f;
}

// B fragment (32x16 bf16) gathered from B-transposed storage bt[col][k]:
// lane col n=lane&15, element j -> K = 16*half + j (contiguous order, same as gather_a)
DEV v16u gather_bt(const unsigned short* base, int stride, int lane) {
  int n = lane & 15, half = lane >> 4;
  const unsigned* rp = (const unsigned*)(base + n * stride);
  v16u f;
  #pragma unroll
  for (int j2 = 0; j2 < 8; ++j2) {
    unsigned d = rp[8 * half + j2];
    f[2*j2]     = (unsigned short)(d & 0xffffu);
    f[2*j2 + 1] = (unsigned short)(d >> 16);
  }
  return f;
}

// Prepacked B fragments: one v16u per lane per (colTile, kBlock)
DEV v16u ld_fragB(const unsigned short* frag, int ct, int kb, int KB, int lane) {
  return ((const v16u*)frag)[(size_t)((ct * KB + kb) * 32 + lane)];
}

// ---------------- prep kernels ----------------

// sup1 = softmax(relu(emb @ emb^T)) -> bf16, padded to 320x320 with zeros
__global__ void k_sup(const float* __restrict__ emb, unsigned short* __restrict__ sup) {
  __shared__ float sc[NP_];
  __shared__ float red[2];
  int n = blockIdx.x, m = threadIdx.x;
  float s = 0.f;
  if (n < N_ && m < N_) {
    #pragma unroll
    for (int e = 0; e < E_; ++e) s += emb[n*E_+e] * emb[m*E_+e];
    s = fmaxf(s, 0.f);
  }
  sc[m] = (m < N_) ? s : -1e30f;
  __syncthreads();
  if (m == 0) { float mx = -1e30f; for (int i = 0; i < N_; ++i) mx = fmaxf(mx, sc[i]); red[0] = mx; }
  __syncthreads();
  float ex = (n < N_ && m < N_) ? __expf(s - red[0]) : 0.f;
  sc[m] = ex;
  __syncthreads();
  if (m == 0) { float sm = 0.f; for (int i = 0; i < N_; ++i) sm += sc[i]; red[1] = sm; }
  __syncthreads();
  sup[(size_t)n*NP_ + m] = (n < N_) ? f2bf(ex / red[1]) : (unsigned short)0;
}

// pack row-major fp32 weight (Kdim x Ncols) into bf16 B-fragment layout
__global__ void k_pack(const float* __restrict__ src, unsigned short* __restrict__ dst,
                       int Kdim, int Ncols) {
  int idx = blockIdx.x * blockDim.x + threadIdx.x;
  if (idx >= Kdim * Ncols) return;
  int k = idx / Ncols, c = idx % Ncols;
  int kb = k >> 5, kk = k & 31, half = kk >> 4, j = kk & 15;
  int ct = c >> 4, nn = c & 15, lane = nn + (half << 4);
  int KB = Kdim >> 5;
  dst[((size_t)((ct * KB + kb) * 32 + lane)) * 16 + j] = f2bf(src[idx]);
}

// gW[n] = emb[n] . weights_pool  -> per-node 128x64 bf16 B-fragments
__global__ void k_gw(const float* __restrict__ emb, const float* __restrict__ pool,
                     unsigned short* __restrict__ gwfrag) {
  int idx = blockIdx.x * blockDim.x + threadIdx.x;     // n*8192 + k*64 + o
  if (idx >= N_ * 128 * 64) return;
  int o = idx & 63, k = (idx >> 6) & 127, n = idx >> 13;
  float s = 0.f;
  #pragma unroll
  for (int e = 0; e < E_; ++e) s += emb[n*E_+e] * pool[(size_t)(e*128 + k)*64 + o];
  int kb = k >> 5, kk = k & 31, half = kk >> 4, j = kk & 15;
  int ct = o >> 4, nn = o & 15, lane = nn + (half << 4);
  gwfrag[(size_t)n*8192 + ((ct*4 + kb)*32 + lane)*16 + j] = f2bf(s);
}

__global__ void k_gb(const float* __restrict__ emb, const float* __restrict__ bpool,
                     float* __restrict__ gb) {
  int idx = blockIdx.x * blockDim.x + threadIdx.x;
  if (idx >= N_ * 64) return;
  int o = idx & 63, n = idx >> 6;
  float s = 0.f;
  #pragma unroll
  for (int e = 0; e < E_; ++e) s += emb[n*E_+e] * bpool[e*64 + o];
  gb[idx] = s;
}

// h0/z0 from X0 (C=2)
__global__ void k_init(const float* __restrict__ a, const float* __restrict__ Wh0,
                       const float* __restrict__ bh0, const float* __restrict__ Wz0,
                       const float* __restrict__ bz0, float* __restrict__ h,
                       float* __restrict__ z) {
  int idx = blockIdx.x * blockDim.x + threadIdx.x;
  if (idx >= M_ * 64) return;
  int hh = idx & 63, row = idx >> 6;
  int n = row >> 6, b = row & 63;
  const float* xp = a + ((size_t)(b*N_ + n)*12)*2;
  float x0 = xp[0], x1 = xp[1];
  h[idx] = x0*Wh0[hh] + x1*Wh0[64+hh] + bh0[hh];
  z[idx] = x0*Wz0[hh] + x1*Wz0[64+hh] + bz0[hh];
}

// ---------------- RK4 glue ----------------

__global__ void k_seed(const float* __restrict__ h, const float* __restrict__ z,
                       unsigned short* __restrict__ hb, unsigned short* __restrict__ zb,
                       float* __restrict__ acch, float* __restrict__ accz) {
  int idx = blockIdx.x * blockDim.x + threadIdx.x;
  if (idx >= M_ * 64) return;
  hb[idx] = f2bf(h[idx]); zb[idx] = f2bf(z[idx]);
  acch[idx] = 0.f; accz[idx] = 0.f;
}

__global__ void k_post(float* __restrict__ h, float* __restrict__ z,
                       const float* __restrict__ dh, const float* __restrict__ dz,
                       float* __restrict__ acch, float* __restrict__ accz,
                       unsigned short* __restrict__ hb, unsigned short* __restrict__ zb,
                       float wa, float cn, int fin) {
  int idx = blockIdx.x * blockDim.x + threadIdx.x;
  if (idx >= M_ * 64) return;
  float kh = dh[idx], kz = dz[idx];
  if (fin) {
    h[idx] += (1.f/6.f) * (acch[idx] + wa*kh);
    z[idx] += (1.f/6.f) * (accz[idx] + wa*kz);
  } else {
    acch[idx] += wa*kh; accz[idx] += wa*kz;
    hb[idx] = f2bf(h[idx] + cn*kh);
    zb[idx] = f2bf(z[idx] + cn*kz);
  }
}

// ---------------- vf kernels (wave = one 16-row tile) ----------------

// f-branch: u=relu(h@Wf1+b); u=relu(u@Wf2+b); Ff=tanh(u@Wf3+b); dh=Ff.dX
__global__ __launch_bounds__(128) void k_f(
    const unsigned short* __restrict__ hb,
    const unsigned short* __restrict__ Wf1, const unsigned short* __restrict__ Wf2,
    const unsigned short* __restrict__ Wf3,
    const float* __restrict__ bin, const float* __restrict__ bmid,
    const float* __restrict__ bout,
    const float* __restrict__ Bi, const float* __restrict__ Ci,
    const float* __restrict__ Di, int tIdx, float frac,
    float* __restrict__ dh) {
  __shared__ unsigned short uT[4][16*66];
  __shared__ float dxT[4][16][2];
  int w = threadIdx.x >> 5, lane = threadIdx.x & 31;
  int nn = lane & 15, half = lane >> 4;
  int row0 = (blockIdx.x * 4 + w) * 16;
  {                                             // dX for the 16 rows, 2 channels
    int m = nn, c = half, row = row0 + m;
    int nb = row >> 6, bb = row & 63;
    size_t off = ((size_t)(bb*N_ + nb)*12 + tIdx)*2 + c;
    dxT[w][m][c] = Bi[off] + (Ci[off] + Di[off]*frac)*frac;
  }
  v16u a0 = gather_a(hb + (size_t)row0*64, 64, lane);
  v16u a1 = gather_a(hb + (size_t)row0*64 + 32, 64, lane);
  #pragma unroll
  for (int ct = 0; ct < 4; ++ct) {
    v8f c = splat8(bin[ct*16 + nn]);
    c = wmma_bf16(a0, ld_fragB(Wf1, ct, 0, 2, lane), c);
    c = wmma_bf16(a1, ld_fragB(Wf1, ct, 1, 2, lane), c);
    #pragma unroll
    for (int r = 0; r < 8; ++r)
      uT[w][(r + 8*half)*66 + ct*16 + nn] = f2bf(fmaxf(c[r], 0.f));
  }
  __syncthreads();
  a0 = gather_a(&uT[w][0], 66, lane);
  a1 = gather_a(&uT[w][32], 66, lane);
  __syncthreads();
  #pragma unroll
  for (int ct = 0; ct < 4; ++ct) {
    v8f c = splat8(bmid[ct*16 + nn]);
    c = wmma_bf16(a0, ld_fragB(Wf2, ct, 0, 2, lane), c);
    c = wmma_bf16(a1, ld_fragB(Wf2, ct, 1, 2, lane), c);
    #pragma unroll
    for (int r = 0; r < 8; ++r)
      uT[w][(r + 8*half)*66 + ct*16 + nn] = f2bf(fmaxf(c[r], 0.f));
  }
  __syncthreads();
  a0 = gather_a(&uT[w][0], 66, lane);
  a1 = gather_a(&uT[w][32], 66, lane);
  #pragma unroll
  for (int ct = 0; ct < 8; ++ct) {
    v8f c = splat8(bout[ct*16 + nn]);
    c = wmma_bf16(a0, ld_fragB(Wf3, ct, 0, 2, lane), c);
    c = wmma_bf16(a1, ld_fragB(Wf3, ct, 1, 2, lane), c);
    #pragma unroll
    for (int r = 0; r < 8; ++r) {
      float t = fast_tanh(c[r]) * dxT[w][r + 8*half][nn & 1];
      float o = t + __shfl_xor(t, 1, 32);       // pair (c=0)+(c=1)
      if ((nn & 1) == 0)
        dh[(size_t)(row0 + r + 8*half)*64 + ((ct*16 + nn) >> 1)] = o;
    }
  }
}

// g1: v = relu(z@g_Win+b); write row-major (xgcat cols 0..63) + transposed vT
__global__ __launch_bounds__(128) void k_g1(
    const unsigned short* __restrict__ zb, const unsigned short* __restrict__ Wg1,
    const float* __restrict__ gbin,
    unsigned short* __restrict__ xgcat, unsigned short* __restrict__ vT) {
  int w = threadIdx.x >> 5, lane = threadIdx.x & 31;
  int nn = lane & 15, half = lane >> 4;
  int row0 = (blockIdx.x * 4 + w) * 16;
  v16u a0 = gather_a(zb + (size_t)row0*64, 64, lane);
  v16u a1 = gather_a(zb + (size_t)row0*64 + 32, 64, lane);
  #pragma unroll
  for (int ct = 0; ct < 4; ++ct) {
    v8f c = splat8(gbin[ct*16 + nn]);
    c = wmma_bf16(a0, ld_fragB(Wg1, ct, 0, 2, lane), c);
    c = wmma_bf16(a1, ld_fragB(Wg1, ct, 1, 2, lane), c);
    int hh = ct*16 + nn;
    #pragma unroll
    for (int r = 0; r < 8; ++r) {
      unsigned short bv = f2bf(fmaxf(c[r], 0.f));
      int row = row0 + r + 8*half;
      xgcat[(size_t)row*128 + hh] = bv;
      int nb = row >> 6, bb = row & 63;
      vT[((size_t)(bb*64 + hh))*NP_ + nb] = bv;  // (b,h)-major, node columns
    }
  }
}

// graph conv: out(320 x 4096) = sup1p @ v  -> xgcat cols 64..127
__global__ __launch_bounds__(128) void k_gc(
    const unsigned short* __restrict__ sup, const unsigned short* __restrict__ vT,
    unsigned short* __restrict__ xgcat) {
  int w = threadIdx.x >> 5, lane = threadIdx.x & 31;
  int nn = lane & 15, half = lane >> 4;
  int job = blockIdx.x * 4 + w;                 // 20 row-tiles x 64 col-groups
  int rt = job >> 6, cg = job & 63;
  v8f acc[4];
  #pragma unroll
  for (int j = 0; j < 4; ++j) acc[j] = splat8(0.f);
  for (int kb = 0; kb < 10; ++kb) {
    v16u a = gather_a(sup + (size_t)(rt*16)*NP_ + kb*32, NP_, lane);
    #pragma unroll
    for (int j = 0; j < 4; ++j) {
      v16u bf = gather_bt(vT + (size_t)((cg*4 + j)*16)*NP_ + kb*32, NP_, lane);
      acc[j] = wmma_bf16(a, bf, acc[j]);
    }
  }
  #pragma unroll
  for (int j = 0; j < 4; ++j) {
    int col = (cg*4 + j)*16 + nn;
    int bb = col >> 6, hh = col & 63;
    #pragma unroll
    for (int r = 0; r < 8; ++r) {
      int n = rt*16 + r + 8*half;
      if (n < N_)
        xgcat[((size_t)(n*64 + bb))*128 + 64 + hh] = f2bf(acc[j][r]);
    }
  }
}

// g2: vv = xgcat@gW[n]+gb ; dz = tanh(vv@g_Wout+b).reshape(H,H) . dh.
// g_Wout fragment stream (8KB per hcol) is staged block-wide into LDS with
// double-buffered global_load_async_to_lds_b128 (ASYNCcnt), shared by 4 waves.
__global__ __launch_bounds__(128) void k_g2(
    const unsigned short* __restrict__ xgcat, const unsigned short* __restrict__ gwfrag,
    const float* __restrict__ gb, const unsigned short* __restrict__ Wg2,
    const float* __restrict__ gbout, const float* __restrict__ dh,
    float* __restrict__ dz) {
  __shared__ unsigned short vvT[4][16*66];
  __shared__ float dzT[4][16*64];
  __shared__ __align__(32) unsigned short wgB[2][4096];   // 2 x 8KB staging
  int w = threadIdx.x >> 5, lane = threadIdx.x & 31;
  int nn = lane & 15, half = lane >> 4;
  int row0 = (blockIdx.x * 4 + w) * 16;
  int node = row0 >> 6;                         // fixed node per 16-row tile

  // kick off async stage of hcol=0 fragments (overlaps with GEMM-7 below)
  {
    int t = threadIdx.x;
    #pragma unroll
    for (int q = 0; q < 4; ++q) {
      int c = t + 128*q;                        // 512 x 16B chunks
      async_ld_b128((unsigned)(size_t)(&wgB[0][c*8]), Wg2 + c*8);
    }
  }

  // GEMM-7: per-node einsum  vv = xgcat @ gW[node] + gb
  v16u a4[4];
  #pragma unroll
  for (int kb = 0; kb < 4; ++kb)
    a4[kb] = gather_a(xgcat + (size_t)row0*128 + kb*32, 128, lane);
  const unsigned short* gw = gwfrag + (size_t)node*8192;
  #pragma unroll
  for (int ct = 0; ct < 4; ++ct) {
    v8f c = splat8(gb[node*64 + ct*16 + nn]);
    #pragma unroll
    for (int kb = 0; kb < 4; ++kb)
      c = wmma_bf16(a4[kb], ld_fragB(gw, ct, kb, 4, lane), c);
    #pragma unroll
    for (int r = 0; r < 8; ++r)
      vvT[w][(r + 8*half)*66 + ct*16 + nn] = f2bf(c[r]);
  }
  __syncthreads();
  v16u a0 = gather_a(&vvT[w][0], 66, lane);
  v16u a1 = gather_a(&vvT[w][32], 66, lane);
  float dhr[4][8];
  #pragma unroll
  for (int it = 0; it < 4; ++it) {
    #pragma unroll
    for (int r = 0; r < 8; ++r)
      dhr[it][r] = dh[(size_t)(row0 + r + 8*half)*64 + it*16 + nn];
  }
  for (int hcol = 0; hcol < 64; ++hcol) {       // Gg columns h*64..h*64+63
    int buf = hcol & 1;
    if (hcol < 63) {                            // prefetch next slab, drain current
      int t = threadIdx.x;
      const unsigned short* src = Wg2 + (size_t)(hcol + 1)*4096;
      #pragma unroll
      for (int q = 0; q < 4; ++q) {
        int c = t + 128*q;
        async_ld_b128((unsigned)(size_t)(&wgB[buf ^ 1][c*8]), src + c*8);
      }
      asm volatile("s_wait_asynccnt 4" ::: "memory");
    } else {
      asm volatile("s_wait_asynccnt 0" ::: "memory");
    }
    __syncthreads();                            // slab visible to all 4 waves
    float part[8];
    #pragma unroll
    for (int r = 0; r < 8; ++r) part[r] = 0.f;
    #pragma unroll
    for (int it = 0; it < 4; ++it) {
      v16u b0 = ((const v16u*)&wgB[buf][0])[(it*2 + 0)*32 + lane];
      v16u b1 = ((const v16u*)&wgB[buf][0])[(it*2 + 1)*32 + lane];
      v8f c = splat8(gbout[(hcol*4 + it)*16 + nn]);
      c = wmma_bf16(a0, b0, c);
      c = wmma_bf16(a1, b1, c);
      #pragma unroll
      for (int r = 0; r < 8; ++r) part[r] += fast_tanh(c[r]) * dhr[it][r];
    }
    #pragma unroll
    for (int off = 1; off < 16; off <<= 1) {
      #pragma unroll
      for (int r = 0; r < 8; ++r) part[r] += __shfl_xor(part[r], off, 32);
    }
    if (nn == 0) {
      #pragma unroll
      for (int r = 0; r < 8; ++r) dzT[w][(r + 8*half)*64 + hcol] = part[r];
    }
    __syncthreads();                            // all waves done before restage
  }
  #pragma unroll
  for (int i = 0; i < 32; ++i)
    dz[(size_t)row0*64 + i*32 + lane] = dzT[w][i*32 + lane];
}

// final projection: out[b,o,n] = z[n,b,:] . Wend[o,:] + bend[o]
__global__ void k_out(const float* __restrict__ z, const float* __restrict__ Wend,
                      const float* __restrict__ bend, float* __restrict__ out) {
  int idx = blockIdx.x * blockDim.x + threadIdx.x;
  if (idx >= B_ * 12 * N_) return;
  int n = idx % N_; int o = (idx / N_) % 12; int bb = idx / (N_ * 12);
  const float* zr = z + (size_t)(n*64 + bb)*64;
  const float* wr = Wend + o*64;
  float s = bend[o];
  #pragma unroll 8
  for (int hh = 0; hh < 64; ++hh) s += zr[hh] * wr[hh];
  out[idx] = s;
}

// ---------------- host ----------------

extern "C" void kernel_launch(void* const* d_in, const int* in_sizes, int n_in,
                              void* d_out, int out_size, void* d_ws, size_t ws_size,
                              hipStream_t stream) {
  (void)in_sizes; (void)n_in; (void)out_size; (void)ws_size;
  const float* In_a  = (const float*)d_in[1];
  const float* In_b  = (const float*)d_in[2];
  const float* In_c  = (const float*)d_in[3];
  const float* In_d  = (const float*)d_in[4];
  const float* Wh0   = (const float*)d_in[5];  const float* bh0  = (const float*)d_in[6];
  const float* Wz0   = (const float*)d_in[7];  const float* bz0  = (const float*)d_in[8];
  const float* fWin  = (const float*)d_in[9];  const float* fbin = (const float*)d_in[10];
  const float* fWmid = (const float*)d_in[11]; const float* fbmid= (const float*)d_in[12];
  const float* fWout = (const float*)d_in[13]; const float* fbout= (const float*)d_in[14];
  const float* gWin  = (const float*)d_in[15]; const float* gbin = (const float*)d_in[16];
  const float* gWout = (const float*)d_in[17]; const float* gbout= (const float*)d_in[18];
  const float* emb   = (const float*)d_in[19]; const float* pool = (const float*)d_in[20];
  const float* bpool = (const float*)d_in[21];
  const float* Wend  = (const float*)d_in[22]; const float* bend = (const float*)d_in[23];

  char* base = (char*)d_ws; size_t off = 0;
  auto alloc = [&](size_t bytes) {
    off = (off + 255) & ~(size_t)255;
    void* r = base + off; off += bytes; return r;
  };
  float* h    = (float*)alloc((size_t)M_*64*4);
  float* z    = (float*)alloc((size_t)M_*64*4);
  float* acch = (float*)alloc((size_t)M_*64*4);
  float* accz = (float*)alloc((size_t)M_*64*4);
  float* dh   = (float*)alloc((size_t)M_*64*4);
  float* dz   = (float*)alloc((size_t)M_*64*4);
  unsigned short* hb    = (unsigned short*)alloc((size_t)M_*64*2);
  unsigned short* zb    = (unsigned short*)alloc((size_t)M_*64*2);
  unsigned short* xgcat = (unsigned short*)alloc((size_t)M_*128*2);
  unsigned short* vT    = (unsigned short*)alloc((size_t)4096*NP_*2);
  unsigned short* sup   = (unsigned short*)alloc((size_t)NP_*NP_*2);
  unsigned short* wf1   = (unsigned short*)alloc(4096*2);
  unsigned short* wf2   = (unsigned short*)alloc(4096*2);
  unsigned short* wf3   = (unsigned short*)alloc(8192*2);
  unsigned short* wg1   = (unsigned short*)alloc(4096*2);
  unsigned short* wg2   = (unsigned short*)alloc((size_t)64*4096*2);
  unsigned short* gwfrag= (unsigned short*)alloc((size_t)N_*8192*2);
  float* gbuf = (float*)alloc((size_t)N_*64*4);

  // one-time preprocessing (recomputed each launch: deterministic)
  k_sup<<<NP_, NP_, 0, stream>>>(emb, sup);
  k_pack<<<(4096 + 255)/256, 256, 0, stream>>>(fWin,  wf1, 64, 64);
  k_pack<<<(4096 + 255)/256, 256, 0, stream>>>(fWmid, wf2, 64, 64);
  k_pack<<<(8192 + 255)/256, 256, 0, stream>>>(fWout, wf3, 64, 128);
  k_pack<<<(4096 + 255)/256, 256, 0, stream>>>(gWin,  wg1, 64, 64);
  k_pack<<<(64*4096 + 255)/256, 256, 0, stream>>>(gWout, wg2, 64, 4096);
  k_gw<<<(N_*128*64 + 255)/256, 256, 0, stream>>>(emb, pool, gwfrag);
  k_gb<<<(N_*64 + 255)/256, 256, 0, stream>>>(emb, bpool, gbuf);
  k_init<<<(M_*64 + 255)/256, 256, 0, stream>>>(In_a, Wh0, bh0, Wz0, bz0, h, z);

  const int EW = (M_*64 + 255)/256;
  for (int i = 0; i < STEPS; ++i) {            // dt = 1 (times = arange)
    k_seed<<<EW, 256, 0, stream>>>(h, z, hb, zb, acch, accz);
    for (int s = 0; s < 4; ++s) {
      int tIdx; float frac;
      if (s == 0)      { tIdx = (i == 0) ? 0 : i - 1; frac = (i == 0) ? 0.f : 1.f; }
      else if (s < 3)  { tIdx = i; frac = 0.5f; }
      else             { tIdx = i; frac = 1.f; }
      k_f <<<N_,  128, 0, stream>>>(hb, wf1, wf2, wf3, fbin, fbmid, fbout,
                                    In_b, In_c, In_d, tIdx, frac, dh);
      k_g1<<<N_,  128, 0, stream>>>(zb, wg1, gbin, xgcat, vT);
      k_gc<<<NP_, 128, 0, stream>>>(sup, vT, xgcat);
      k_g2<<<N_,  128, 0, stream>>>(xgcat, gwfrag, gbuf, wg2, gbout, dh, dz);
      float wa = (s == 1 || s == 2) ? 2.f : 1.f;
      float cn = (s <= 1) ? 0.5f : 1.f;
      k_post<<<EW, 256, 0, stream>>>(h, z, dh, dz, acch, accz, hb, zb,
                                     wa, cn, (s == 3) ? 1 : 0);
    }
  }
  k_out<<<(B_*12*N_ + 255)/256, 256, 0, stream>>>(z, Wend, bend, (float*)d_out);
}
